// Transformer_48601849921957
// MI455X (gfx1250) — compile-verified, matrix operand order checked by reading, not run on hardware
//
#include <hip/hip_runtime.h>
#include <hip/hip_bf16.h>

typedef __bf16 bf16;
typedef __attribute__((ext_vector_type(16))) __bf16 v16bf;
typedef __attribute__((ext_vector_type(8)))  __bf16 v8bf;
typedef __attribute__((ext_vector_type(8)))  float  v8f;
typedef __attribute__((ext_vector_type(4)))  unsigned int v4u;
typedef __attribute__((ext_vector_type(8)))  int v8i;
typedef __attribute__((ext_vector_type(4)))  int v4i;

#define NLAYER 4
#define Dm     512
#define NHh    8
#define DHd    64
#define BATCH  2
#define Ll     1024
#define Mtok   (BATCH * Ll)   // 2048

#if defined(__AMDGCN__) && __has_builtin(__builtin_amdgcn_tensor_load_to_lds) && \
    __has_builtin(__builtin_amdgcn_s_wait_tensorcnt)
#define HAVE_TDM 1
#else
#define HAVE_TDM 0
#endif

// ---------------------------------------------------------------------------
// WMMA helpers
// ---------------------------------------------------------------------------
__device__ __forceinline__ v8f vzero8() {
  v8f z;
#pragma unroll
  for (int i = 0; i < 8; ++i) z[i] = 0.0f;
  return z;
}

__device__ __forceinline__ v8f wmma_bf16(v16bf a, v16bf b, v8f c) {
  // D = A(16x32) * B(32x16) + C(16x16 f32)
  return __builtin_amdgcn_wmma_f32_16x16x32_bf16(false, a, false, b, (short)0, c,
                                                 false, false);
}

// Fragment loader for BOTH A and B operands of v_wmma_*_16x16x32_bf16.
// Source must be row-major [16 x K] with the "stationary" dim as rows
// (A: M rows; B: rows of B^T). Lane (16*hi+r) holds row r, K runs
// [8*hi,8*hi+8) and [16+8*hi,16+8*hi+8): two 16B loads. ld % 8 == 0.
__device__ __forceinline__ v16bf load_frag(const bf16* base, int row, int ld,
                                           int kBase, int lane) {
  const bf16* p = base + (size_t)(row + (lane & 15)) * ld + kBase + ((lane >> 4) << 3);
  v8bf lo = *(const v8bf*)p;
  v8bf hi = *(const v8bf*)(p + 16);
  v16bf f;
#pragma unroll
  for (int i = 0; i < 8; ++i) { f[i] = lo[i]; f[i + 8] = hi[i]; }
  return f;
}

// ---------------------------------------------------------------------------
// Tensor Data Mover: async 2D tile (bf16) global -> LDS with row padding.
// D# per CDNA5 ISA: count=1, type=2, data_size=2B,
// pad_enable; tile0 = elems/row (K), tile1 = rows; stride0 in elements.
// pad_interval code c -> pad after 2^(c+1) DWORDs; pad_amount code a -> a+1 DWORDs.
// This toolchain exposes the 6-arg builtin (g0, g1, g2, g3, g_extra, cpol).
// ---------------------------------------------------------------------------
#if HAVE_TDM
__device__ __forceinline__ void tdm_load_2d(unsigned lds_addr, const void* gptr,
                                            unsigned tile0, unsigned tile1,
                                            unsigned stride0,
                                            unsigned pad_interval,
                                            unsigned pad_amount) {
  unsigned long long ga = (unsigned long long)gptr;
  v4u g0;
  g0[0] = 1u;                                              // count=1, user mode
  g0[1] = lds_addr;                                        // LDS byte address
  g0[2] = (unsigned)(ga & 0xffffffffull);                  // global addr [31:0]
  g0[3] = (unsigned)((ga >> 32) & 0x1ffffffull) | (2u << 30);  // [56:32], type=2
  v8i g1;
  g1[0] = (int)((1u << 16)                // data_size = 2 bytes
              | (1u << 20)                // pad_enable
              | (pad_interval << 22)
              | (pad_amount << 25));
  g1[1] = (int)((tile0 & 0xffffu) << 16);                        // tensor_dim0 lo16
  g1[2] = (int)((tile0 >> 16) | ((tile1 & 0xffffu) << 16));      // dim0 hi | dim1 lo
  g1[3] = (int)((tile1 >> 16) | (tile0 << 16));                  // dim1 hi | tile_dim0
  g1[4] = (int)(tile1 & 0xffffu);                                // tile_dim1 (dim2=0)
  g1[5] = (int)stride0;                                          // dim0_stride [31:0]
  g1[6] = 0;                                                     // stride hi, dim1_stride lo
  g1[7] = 0;
  v4i z4; z4[0] = 0; z4[1] = 0; z4[2] = 0; z4[3] = 0;            // groups 2/3 unused (2D)
  v8i z8;
#pragma unroll
  for (int i = 0; i < 8; ++i) z8[i] = 0;
  __builtin_amdgcn_tensor_load_to_lds(g0, g1, z4, z4, z8, 0);
}
#endif

// ---------------------------------------------------------------------------
// f32 -> bf16 conversion
// ---------------------------------------------------------------------------
__global__ __launch_bounds__(256) void f2bf_k(const float* __restrict__ in,
                                              bf16* __restrict__ out, int n) {
  int i = blockIdx.x * 256 + threadIdx.x;
  if (i < n) out[i] = (bf16)in[i];
}

// ---------------------------------------------------------------------------
// S = (tril(LU,-1)+I) @ (triu(LU,1)+softplus(diag)I), stored transposed bf16:
// St[h][e][d] = S[h][d][e]
// ---------------------------------------------------------------------------
__global__ __launch_bounds__(64) void buildS_k(const float* __restrict__ LU,
                                               bf16* __restrict__ St) {
  const int h = blockIdx.x, d = threadIdx.x;
  const float* Lh = LU + h * 4096;
  for (int e = 0; e < 64; ++e) {
    int kmax = (d < e) ? d : e;
    float s = 0.0f;
    for (int k = 0; k <= kmax; ++k) {
      float lm = (k < d) ? Lh[d * 64 + k] : 1.0f;
      float um;
      if (k < e) um = Lh[k * 64 + e];
      else { float dg = Lh[k * 64 + k]; um = (dg > 20.f) ? dg : log1pf(expf(dg)); }
      s += lm * um;
    }
    St[(size_t)h * 4096 + e * 64 + d] = (bf16)s;
  }
}

// ---------------------------------------------------------------------------
// Row RMSNorm (512 per row). 256 threads, 2 elems each; wave-shuffle + LDS.
// ---------------------------------------------------------------------------
__global__ __launch_bounds__(256) void rmsnorm_k(const float* __restrict__ x,
                                                 const float* __restrict__ w,
                                                 bf16* __restrict__ outb,
                                                 float* __restrict__ outf,
                                                 float eps) {
  const int row = blockIdx.x, t = threadIdx.x;
  float v0 = x[(size_t)row * Dm + t];
  float v1 = x[(size_t)row * Dm + t + 256];
  float ss = v0 * v0 + v1 * v1;
#pragma unroll
  for (int o = 16; o > 0; o >>= 1) ss += __shfl_xor(ss, o, 32);
  __shared__ float red[8];
  if ((t & 31) == 0) red[t >> 5] = ss;
  __syncthreads();
  float tot = 0.0f;
#pragma unroll
  for (int i = 0; i < 8; ++i) tot += red[i];
  float s = rsqrtf(tot / (float)Dm + eps);
  float r0 = v0 * s * w[t], r1 = v1 * s * w[t + 256];
  outb[(size_t)row * Dm + t]       = (bf16)r0;
  outb[(size_t)row * Dm + t + 256] = (bf16)r1;
  if (outf) {
    outf[(size_t)row * Dm + t]       = r0;
    outf[(size_t)row * Dm + t + 256] = r1;
  }
}

// ---------------------------------------------------------------------------
// Per-head RMSNorm over 64-dim heads (eps 1e-8). One wave per head per row.
// ---------------------------------------------------------------------------
__global__ __launch_bounds__(256) void mhnorm_k(const float* __restrict__ in,
                                                const float* __restrict__ w,
                                                bf16* __restrict__ out) {
  const int row = blockIdx.x, t = threadIdx.x;
  const int h = t >> 5, e = t & 31;
  const size_t base = (size_t)row * Dm + h * 64 + e;
  float v0 = in[base], v1 = in[base + 32];
  float ss = v0 * v0 + v1 * v1;
#pragma unroll
  for (int o = 16; o > 0; o >>= 1) ss += __shfl_xor(ss, o, 32);
  float s = rsqrtf(ss / 64.0f + 1e-8f);
  out[base]      = (bf16)(v0 * s * w[h * 64 + e]);
  out[base + 32] = (bf16)(v1 * s * w[h * 64 + e + 32]);
}

// ---------------------------------------------------------------------------
// Tiled WMMA GEMM: C(MxN) = A(MxK) * W^T, W row-major (N,K) bf16.
// Block tile 128x128, BK=32, 8 waves, wave tile 32x64 (2x4 WMMA tiles).
// TDM double-buffered LDS staging (wave 0 drives the DMA, TENSORcnt sync).
// MODE 0: Cb = bf16(gelu(acc));  MODE 1: resid += acc + bias[n];  MODE 2: Cf = acc
// ---------------------------------------------------------------------------
template <int MODE>
__global__ __launch_bounds__(256) void gemm_k(const bf16* __restrict__ A, int lda,
                                              const bf16* __restrict__ Bw, int ldb,
                                              int K,
                                              float* __restrict__ Cf,
                                              bf16* __restrict__ Cb, int ldc,
                                              const float* __restrict__ bias,
                                              float* __restrict__ resid) {
  const int tid = threadIdx.x, lane = tid & 31, wv = tid >> 5;
  const int wm = wv >> 1, wn = wv & 1;
  const int mBase = blockIdx.y * 128, nBase = blockIdx.x * 128;
  __shared__ bf16 As[2][128 * 40];
  __shared__ bf16 Bs[2][128 * 40];
  v8f acc[2][4];
#pragma unroll
  for (int i = 0; i < 2; ++i)
#pragma unroll
    for (int j = 0; j < 4; ++j) acc[i][j] = vzero8();

#if HAVE_TDM
  if (wv == 0) {  // prime buffer 0 (row = 32 bf16 = 16 DW; pad 4 DW -> pitch 40)
    tdm_load_2d((unsigned)(size_t)&As[0][0], A + (size_t)mBase * lda, 32, 128,
                (unsigned)lda, 3, 3);
    tdm_load_2d((unsigned)(size_t)&Bs[0][0], Bw + (size_t)nBase * ldb, 32, 128,
                (unsigned)ldb, 3, 3);
  }
#endif

  int ib = 0;
  for (int kk = 0; kk < K; kk += 32, ib ^= 1) {
#if HAVE_TDM
    if (wv == 0) __builtin_amdgcn_s_wait_tensorcnt(0);
    __syncthreads();
    if (wv == 0 && kk + 32 < K) {  // DMA next panels while computing this one
      tdm_load_2d((unsigned)(size_t)&As[ib ^ 1][0],
                  A + (size_t)mBase * lda + kk + 32, 32, 128, (unsigned)lda, 3, 3);
      tdm_load_2d((unsigned)(size_t)&Bs[ib ^ 1][0],
                  Bw + (size_t)nBase * ldb + kk + 32, 32, 128, (unsigned)ldb, 3, 3);
    }
#else
#pragma unroll
    for (int i = 0; i < 2; ++i) {
      int c = tid + (i << 8);
      int r = c >> 2, col = (c & 3) << 3;
      *(v8bf*)&As[ib][r * 40 + col] = *(const v8bf*)&A[(size_t)(mBase + r) * lda + kk + col];
      *(v8bf*)&Bs[ib][r * 40 + col] = *(const v8bf*)&Bw[(size_t)(nBase + r) * ldb + kk + col];
    }
    __syncthreads();
#endif
    const bf16* Asp = As[ib];
    const bf16* Bsp = Bs[ib];
    v16bf af[2], bfr[4];
#pragma unroll
    for (int i = 0; i < 2; ++i) af[i] = load_frag(Asp, wm * 32 + i * 16, 40, 0, lane);
#pragma unroll
    for (int j = 0; j < 4; ++j) bfr[j] = load_frag(Bsp, wn * 64 + j * 16, 40, 0, lane);
#pragma unroll
    for (int j = 0; j < 4; ++j)
#pragma unroll
      for (int i = 0; i < 2; ++i) acc[i][j] = wmma_bf16(af[i], bfr[j], acc[i][j]);
    __syncthreads();
  }

  const int cl = lane & 15, ch = lane >> 4;
#pragma unroll
  for (int i = 0; i < 2; ++i)
#pragma unroll
    for (int j = 0; j < 4; ++j) {
      const int m0 = mBase + wm * 32 + i * 16 + ch * 8;
      const int n  = nBase + wn * 64 + j * 16 + cl;
#pragma unroll
      for (int r = 0; r < 8; ++r) {
        float v = acc[i][j][r];
        const size_t idx = (size_t)(m0 + r) * ldc + n;
        if (MODE == 0) {
          float g = 0.5f * v * (1.0f + erff(v * 0.70710678118654752f));
          Cb[idx] = (bf16)g;
        } else if (MODE == 1) {
          resid[idx] += v + bias[n];
        } else {
          Cf[idx] = v;
        }
      }
    }
}

// ---------------------------------------------------------------------------
// Chunked causal linear attention, one workgroup per (b,h), 4 waves.
// Per 64-chunk: A=tril(QK^T); xa = A@V + Q@S; S^T += V^T K (f32 WMMA frags).
// Q/K chunk staged by TDM; transposed K^T/V^T copies written by VALU.
// ---------------------------------------------------------------------------
__global__ __launch_bounds__(128) void linatt_k(const bf16* __restrict__ Qg,
                                                const bf16* __restrict__ Kg,
                                                const bf16* __restrict__ Vg,
                                                bf16* __restrict__ xnext,
                                                float* __restrict__ attn,
                                                int accumulate) {
  const int b = blockIdx.x >> 3, h = blockIdx.x & 7;
  const int tid = threadIdx.x, lane = tid & 31, wv = tid >> 5;
  const int cl = lane & 15, ch = lane >> 4;
  __shared__ bf16 Qc[64 * 72];   // Q chunk (token x d)
  __shared__ bf16 Kc[64 * 72];   // K chunk (token x d)
  __shared__ bf16 Kt[64 * 72];   // K^T     (d x token)
  __shared__ bf16 Vt[64 * 72];   // V^T     (e x token)
  __shared__ bf16 Am[64 * 72];   // masked attention (token x token)
  __shared__ bf16 STb[64 * 72];  // S^T bf16 (e x d)

  v8f st[4];                     // wave wv owns S^T rows [16wv,16wv+16), f32
#pragma unroll
  for (int n = 0; n < 4; ++n) st[n] = vzero8();

  const size_t base = (size_t)b * Ll * Dm + h * 64;

  for (int c = 0; c < 16; ++c) {
    const int j0 = c << 6;
#if HAVE_TDM
    if (wv == 0) {  // row = 64 bf16 = 32 DW; pad 4 DW -> pitch 72 halves
      tdm_load_2d((unsigned)(size_t)&Qc[0], Qg + base + (size_t)j0 * Dm, 64, 64,
                  (unsigned)Dm, 4, 3);
      tdm_load_2d((unsigned)(size_t)&Kc[0], Kg + base + (size_t)j0 * Dm, 64, 64,
                  (unsigned)Dm, 4, 3);
    }
    for (int i = tid; i < 4096; i += 128) {
      const int j = i >> 6, e = i & 63;
      const size_t g = base + (size_t)(j0 + j) * Dm + e;
      bf16 k = Kg[g], v = Vg[g];
      Kt[e * 72 + j] = k;
      Vt[e * 72 + j] = v;
    }
#else
    for (int i = tid; i < 4096; i += 128) {
      const int j = i >> 6, e = i & 63;
      const size_t g = base + (size_t)(j0 + j) * Dm + e;
      bf16 q = Qg[g], k = Kg[g], v = Vg[g];
      Qc[j * 72 + e] = q;
      Kc[j * 72 + e] = k;
      Kt[e * 72 + j] = k;
      Vt[e * 72 + j] = v;
    }
#endif
    // publish current (pre-chunk) state as bf16
#pragma unroll
    for (int n = 0; n < 4; ++n)
#pragma unroll
      for (int r = 0; r < 8; ++r)
        STb[(wv * 16 + ch * 8 + r) * 72 + n * 16 + cl] = (bf16)st[n][r];
#if HAVE_TDM
    if (wv == 0) __builtin_amdgcn_s_wait_tensorcnt(0);
#endif
    __syncthreads();

    // A = Q K^T (rows [16wv,16wv+16)), causal mask j<=i, store bf16
    v16bf q0 = load_frag(Qc, wv * 16, 72, 0, lane);
    v16bf q1 = load_frag(Qc, wv * 16, 72, 32, lane);
#pragma unroll
    for (int tn = 0; tn < 4; ++tn) {
      v8f a = vzero8();
      v16bf k0 = load_frag(Kc, tn * 16, 72, 0, lane);
      v16bf k1 = load_frag(Kc, tn * 16, 72, 32, lane);
      a = wmma_bf16(q0, k0, a);
      a = wmma_bf16(q1, k1, a);
#pragma unroll
      for (int r = 0; r < 8; ++r) {
        const int m = wv * 16 + ch * 8 + r, n = tn * 16 + cl;
        Am[m * 72 + n] = (n <= m) ? (bf16)a[r] : (bf16)0.0f;
      }
    }
    __syncthreads();

    // xa = Am @ V + Q @ S  (output rows [16wv,16wv+16), cols e)
    v16bf am0 = load_frag(Am, wv * 16, 72, 0, lane);
    v16bf am1 = load_frag(Am, wv * 16, 72, 32, lane);
#pragma unroll
    for (int te = 0; te < 4; ++te) {
      v8f o = vzero8();
      v16bf f0 = load_frag(Vt, te * 16, 72, 0, lane);
      v16bf f1 = load_frag(Vt, te * 16, 72, 32, lane);
      v16bf f2 = load_frag(STb, te * 16, 72, 0, lane);
      v16bf f3 = load_frag(STb, te * 16, 72, 32, lane);
      o = wmma_bf16(am0, f0, o);
      o = wmma_bf16(am1, f1, o);
      o = wmma_bf16(q0, f2, o);
      o = wmma_bf16(q1, f3, o);
#pragma unroll
      for (int r = 0; r < 8; ++r) {
        const int m = wv * 16 + ch * 8 + r, e = te * 16 + cl;
        const size_t g = base + (size_t)(j0 + m) * Dm + e;
        xnext[g] = (bf16)o[r];
        if (m & 1) {  // odd global token (j0 even)
          const size_t ai = (size_t)(b * (Ll / 2) + ((j0 + m) >> 1)) * Dm + h * 64 + e;
          if (accumulate) attn[ai] += o[r]; else attn[ai] = o[r];
        }
      }
    }

    // S^T += V^T K  (rows e in [16wv,16wv+16), cols d)
    v16bf ve0 = load_frag(Vt, wv * 16, 72, 0, lane);
    v16bf ve1 = load_frag(Vt, wv * 16, 72, 32, lane);
#pragma unroll
    for (int tn = 0; tn < 4; ++tn) {
      v16bf kt0 = load_frag(Kt, tn * 16, 72, 0, lane);
      v16bf kt1 = load_frag(Kt, tn * 16, 72, 32, lane);
      st[tn] = wmma_bf16(ve0, kt0, st[tn]);
      st[tn] = wmma_bf16(ve1, kt1, st[tn]);
    }
    __syncthreads();
  }
}

// ---------------------------------------------------------------------------
// out = x_orig_odd + attn_res @ S per head. One WG per (b,h,64-row chunk).
// ---------------------------------------------------------------------------
__global__ __launch_bounds__(128) void outk(const float* __restrict__ attn,
                                            const bf16* __restrict__ St,
                                            const float* __restrict__ xorig,
                                            float* __restrict__ out) {
  const int b = blockIdx.x >> 6, h = (blockIdx.x >> 3) & 7, cb = blockIdx.x & 7;
  const int tid = threadIdx.x, lane = tid & 31, wv = tid >> 5;
  const int cl = lane & 15, ch = lane >> 4;
  const int l0 = cb << 6;
  __shared__ bf16 Ac[64 * 72];
  for (int i = tid; i < 4096; i += 128) {
    const int j = i >> 6, d = i & 63;
    Ac[j * 72 + d] = (bf16)attn[(size_t)(b * (Ll / 2) + l0 + j) * Dm + h * 64 + d];
  }
  __syncthreads();
  const bf16* Sh = St + (size_t)h * 4096;  // S^T (e x d), ld=64, 16B-aligned rows
  v16bf a0 = load_frag(Ac, wv * 16, 72, 0, lane);
  v16bf a1 = load_frag(Ac, wv * 16, 72, 32, lane);
#pragma unroll
  for (int te = 0; te < 4; ++te) {
    v8f o = vzero8();
    v16bf s0 = load_frag(Sh, te * 16, 64, 0, lane);
    v16bf s1 = load_frag(Sh, te * 16, 64, 32, lane);
    o = wmma_bf16(a0, s0, o);
    o = wmma_bf16(a1, s1, o);
#pragma unroll
    for (int r = 0; r < 8; ++r) {
      const int m = wv * 16 + ch * 8 + r, e = te * 16 + cl;
      const int lh = l0 + m;
      out[(size_t)(b * (Ll / 2) + lh) * Dm + h * 64 + e] =
          xorig[(size_t)(b * Ll + 2 * lh + 1) * Dm + h * 64 + e] + o[r];
    }
  }
}

// ---------------------------------------------------------------------------
// Host orchestration
// ---------------------------------------------------------------------------
extern "C" void kernel_launch(void* const* d_in, const int* in_sizes, int n_in,
                              void* d_out, int out_size, void* d_ws, size_t ws_size,
                              hipStream_t stream) {
  const float* x_in     = (const float*)d_in[0];
  const float* qW       = (const float*)d_in[1];
  const float* vW       = (const float*)d_in[2];
  const float* qnormW   = (const float*)d_in[3];
  const float* vnormW   = (const float*)d_in[4];
  const float* lnW      = (const float*)d_in[5];
  const float* fcW      = (const float*)d_in[6];
  const float* projW    = (const float*)d_in[7];
  const float* projB    = (const float*)d_in[8];
  const float* varnormW = (const float*)d_in[9];
  const float* LU       = (const float*)d_in[10];
  float* out = (float*)d_out;

  char* ws = (char*)d_ws;
  size_t off = 0;
  auto alloc = [&](size_t bytes) { char* p = ws + off; off += (bytes + 255) & ~(size_t)255; return p; };
  bf16*  fcWb   = (bf16*)alloc((size_t)NLAYER * 4 * Dm * Dm * 2);   // (4,2048,512)
  bf16*  projWb = (bf16*)alloc((size_t)NLAYER * 4 * Dm * Dm * 2);   // (4,512,2048)
  bf16*  qWb    = (bf16*)alloc((size_t)NLAYER * Dm * Dm * 2);
  bf16*  vWb    = (bf16*)alloc((size_t)NLAYER * Dm * Dm * 2);
  bf16*  Stb    = (bf16*)alloc((size_t)NHh * DHd * DHd * 2);
  float* xbuf   = (float*)alloc((size_t)Mtok * Dm * 4);
  bf16*  xn     = (bf16*)alloc((size_t)Mtok * Dm * 2);
  bf16*  hbuf   = (bf16*)alloc((size_t)Mtok * 4 * Dm * 2);          // (2048,2048)
  float* xorigf = (float*)alloc((size_t)Mtok * Dm * 4);
  bf16*  xorigb = (bf16*)alloc((size_t)Mtok * Dm * 2);
  float* tmp    = (float*)alloc((size_t)Mtok * Dm * 4);
  bf16*  Qb     = (bf16*)alloc((size_t)Mtok * Dm * 2);
  bf16*  Vb     = (bf16*)alloc((size_t)Mtok * Dm * 2);
  bf16*  xcur   = (bf16*)alloc((size_t)Mtok * Dm * 2);
  float* attn   = (float*)alloc((size_t)BATCH * (Ll / 2) * Dm * 4);

  // Weight conversions + S build
  {
    int n1 = NLAYER * 4 * Dm * Dm;  // fc / proj
    int n2 = NLAYER * Dm * Dm;      // q / v
    f2bf_k<<<(n1 + 255) / 256, 256, 0, stream>>>(fcW, fcWb, n1);
    f2bf_k<<<(n1 + 255) / 256, 256, 0, stream>>>(projW, projWb, n1);
    f2bf_k<<<(n2 + 255) / 256, 256, 0, stream>>>(qW, qWb, n2);
    f2bf_k<<<(n2 + 255) / 256, 256, 0, stream>>>(vW, vWb, n2);
  }
  buildS_k<<<NHh, 64, 0, stream>>>(LU, Stb);
  (void)hipMemcpyAsync(xbuf, x_in, (size_t)Mtok * Dm * 4, hipMemcpyDeviceToDevice, stream);

  // MLP pre-layers
  for (int i = 0; i < NLAYER; ++i) {
    rmsnorm_k<<<Mtok, 256, 0, stream>>>(xbuf, lnW + i * Dm, xn, nullptr, 1e-5f);
    gemm_k<0><<<dim3(16, 16), 256, 0, stream>>>(
        xn, Dm, fcWb + (size_t)i * 4 * Dm * Dm, Dm, Dm,
        nullptr, hbuf, 4 * Dm, nullptr, nullptr);
    gemm_k<1><<<dim3(4, 16), 256, 0, stream>>>(
        hbuf, 4 * Dm, projWb + (size_t)i * 4 * Dm * Dm, 4 * Dm, 4 * Dm,
        nullptr, nullptr, Dm, projB + i * Dm, xbuf);
  }

  // x_orig = rmsnorm(x, varnormW)
  rmsnorm_k<<<Mtok, 256, 0, stream>>>(xbuf, varnormW, xorigb, xorigf, 1e-5f);

  // Attention layers
  for (int i = 0; i < NLAYER; ++i) {
    gemm_k<2><<<dim3(4, 16), 256, 0, stream>>>(
        xorigb, Dm, qWb + (size_t)i * Dm * Dm, Dm, Dm,
        tmp, nullptr, Dm, nullptr, nullptr);
    mhnorm_k<<<Mtok, 256, 0, stream>>>(tmp, qnormW + i * Dm, Qb);
    gemm_k<2><<<dim3(4, 16), 256, 0, stream>>>(
        xorigb, Dm, vWb + (size_t)i * Dm * Dm, Dm, Dm,
        tmp, nullptr, Dm, nullptr, nullptr);
    mhnorm_k<<<Mtok, 256, 0, stream>>>(tmp, vnormW + i * Dm, Vb);
    linatt_k<<<BATCH * NHh, 128, 0, stream>>>(
        Qb, (i == 0) ? xorigb : xcur, Vb, xcur, attn, (i == 0) ? 0 : 1);
  }

  // out = x_orig_odd + attn_res @ S
  outk<<<BATCH * NHh * 8, 128, 0, stream>>>(attn, Stb, xorigf, out);
}
